// SelfAttention_35596688949361
// MI455X (gfx1250) — matrix operand order repaired and checked by reading, not compiled, not run on hardware
//
#include <hip/hip_runtime.h>
#include <hip/hip_bf16.h>
#include <math.h>

// ---------------------------------------------------------------------------
// Self-attention (B=4, S=2048, D=1024) for MI455X / gfx1250 (wave32, WMMA).
// Pipeline: f32->bf16 converts; qkv = X@W_in^T+b (WMMA); V transpose;
// per batch: scores = Q@K^T (WMMA, f32 out) -> softmax(mask, /sqrt(D)) -> bf16
// probs; O = probs@V (WMMA); out = O@W_out^T + b_out (WMMA, f32 out).
// All GEMMs use v_wmma_f32_16x16x32_bf16 with f32 accumulation, with an
// explicit double-buffered (ping-pong) K pipeline so fragment loads for the
// next 32-wide K slab are in flight while the current slab's WMMAs execute.
// ---------------------------------------------------------------------------

typedef __attribute__((ext_vector_type(16))) __bf16 v16bf;
typedef __attribute__((ext_vector_type(8)))  __bf16 v8bf;
typedef __attribute__((ext_vector_type(8)))  float  v8f;

#define DIMN  1024
#define BATCH 4
#define SEQ   2048

// ---------------------------------------------------------------------------
// f32 -> bf16 pack, 8 elements/thread (32B in, 16B out)
// ---------------------------------------------------------------------------
__global__ __launch_bounds__(256) void cvt_f32_to_bf16(
    const float* __restrict__ src, __bf16* __restrict__ dst, int n8)
{
    int i = blockIdx.x * blockDim.x + threadIdx.x;
    if (i >= n8) return;
    const float4* s = (const float4*)src;
    float4 a = s[2 * i];
    float4 b = s[2 * i + 1];
    v8bf o;
    o[0] = (__bf16)a.x; o[1] = (__bf16)a.y; o[2] = (__bf16)a.z; o[3] = (__bf16)a.w;
    o[4] = (__bf16)b.x; o[5] = (__bf16)b.y; o[6] = (__bf16)b.z; o[7] = (__bf16)b.w;
    *(v8bf*)(dst + (size_t)i * 8) = o;
}

// ---------------------------------------------------------------------------
// Fragment loaders following CDNA5 ISA 16-bit A(16x32)/B(32x16) lane layouts.
// A: lanes 0-15: M=lane, elems 0..7 = K0..7, 8..15 = K16..23
//    lanes 16-31: M=lane-16, elems 0..7 = K8..15, 8..15 = K24..31
// B: same K order as A (lanes 0-15: K0..7,K16..23; lanes 16-31: K8..15,K24..31)
// ---------------------------------------------------------------------------
__device__ __forceinline__ void load_frags(
    const __bf16* __restrict__ A, int lda,
    const __bf16* __restrict__ B, int ldb,
    int arow, int akhalf, int bcol0, int bkhalf, int kb,
    v16bf (&af)[4], v16bf (&bf)[2])
{
    #pragma unroll
    for (int i = 0; i < 4; ++i) {
        const __bf16* pa = A + (size_t)(arow + 16 * i) * lda + (kb + akhalf);
        v8bf lo = *(const v8bf*)pa;          // K = kb+akhalf .. +7
        v8bf hi = *(const v8bf*)(pa + 16);   // K = kb+akhalf+16 .. +23
        af[i] = __builtin_shufflevector(lo, hi,
            0, 1, 2, 3, 4, 5, 6, 7, 8, 9, 10, 11, 12, 13, 14, 15);
    }
    #pragma unroll
    for (int j = 0; j < 2; ++j) {
        const __bf16* pb = B + (size_t)(bcol0 + 16 * j) * ldb + (kb + bkhalf);
        v8bf blo = *(const v8bf*)pb;
        v8bf bhi = *(const v8bf*)(pb + 16);
        bf[j] = __builtin_shufflevector(blo, bhi,
            0, 1, 2, 3, 4, 5, 6, 7, 8, 9, 10, 11, 12, 13, 14, 15);
    }
}

// ---------------------------------------------------------------------------
// WMMA GEMM: C[M,N] = A[M,K] * B[N,K]^T (+ bias[N]).
// A, B: bf16 row-major with leading dims lda/ldb (elements), K-contiguous.
// Block: 256 threads = 8 waves, 128x128 tile; wave grid 2(M) x 4(N),
// each wave computes 64x32 via a 4x2 grid of 16x16x32 WMMA tiles.
// Requires K a multiple of 64 and K >= 64 (true for all call sites).
// ---------------------------------------------------------------------------
template <bool OUT_BF16, bool HAS_BIAS>
__global__ __launch_bounds__(256) void gemm_bf16_wmma(
    const __bf16* __restrict__ A, int lda,
    const __bf16* __restrict__ B, int ldb,
    const float* __restrict__ bias,
    void* __restrict__ Cout, int ldc,
    int K)
{
    const int lane  = threadIdx.x & 31;
    const int wave  = threadIdx.x >> 5;
    const int waveM = wave >> 2;                       // 0..1
    const int waveN = wave & 3;                        // 0..3
    const int mWave = blockIdx.y * 128 + waveM * 64;   // wave's M base
    const int nWave = blockIdx.x * 128 + waveN * 32;   // wave's N base

    const int l15 = lane & 15;
    const int lhi = lane >> 4;
    const int arow   = mWave + l15;
    const int akhalf = lhi << 3;     // 0 or 8
    const int bcol0  = nWave + l15;
    const int bkhalf = lhi << 3;     // 0 or 8

    v8f acc[4][2] = {};
    v16bf a0[4], a1[4];
    v16bf b0[2], b1[2];

#define MMA_STAGE(AF, BF)                                                     \
    do {                                                                      \
        _Pragma("unroll")                                                     \
        for (int i = 0; i < 4; ++i)                                           \
            _Pragma("unroll")                                                 \
            for (int j = 0; j < 2; ++j)                                       \
                acc[i][j] = __builtin_amdgcn_wmma_f32_16x16x32_bf16(          \
                    false, AF[i], false, BF[j],                               \
                    (short)0, acc[i][j], false, false);                       \
    } while (0)

    // Prologue: stage 0 in flight.
    load_frags(A, lda, B, ldb, arow, akhalf, bcol0, bkhalf, 0, a0, b0);

    // Steady state: while WMMAs consume one buffer, loads fill the other.
    int kb = 0;
    for (; kb + 64 < K; kb += 64) {
        __builtin_prefetch(A + (size_t)arow * lda + (kb + 96 + akhalf));
        __builtin_prefetch(B + (size_t)bcol0 * ldb + (kb + 96 + bkhalf));
        load_frags(A, lda, B, ldb, arow, akhalf, bcol0, bkhalf, kb + 32, a1, b1);
        MMA_STAGE(a0, b0);
        load_frags(A, lda, B, ldb, arow, akhalf, bcol0, bkhalf, kb + 64, a0, b0);
        MMA_STAGE(a1, b1);
    }
    // Epilogue: kb == K - 64; last two stages, no overshooting loads.
    load_frags(A, lda, B, ldb, arow, akhalf, bcol0, bkhalf, kb + 32, a1, b1);
    MMA_STAGE(a0, b0);
    MMA_STAGE(a1, b1);
#undef MMA_STAGE

    // C layout: VGPR r -> rows (r, r+8) for lane halves; N = lane&15
    float*  Cf = (float*)Cout;
    __bf16* Cb = (__bf16*)Cout;
    #pragma unroll
    for (int i = 0; i < 4; ++i) {
        #pragma unroll
        for (int j = 0; j < 2; ++j) {
            const int col  = nWave + 16 * j + l15;
            const int row0 = mWave + 16 * i + (lhi << 3);
            float bv = 0.0f;
            if (HAS_BIAS) bv = bias[col];
            #pragma unroll
            for (int r = 0; r < 8; ++r) {
                float v = acc[i][j][r] + bv;
                if (OUT_BF16) Cb[(size_t)(row0 + r) * ldc + col] = (__bf16)v;
                else          Cf[(size_t)(row0 + r) * ldc + col] = v;
            }
        }
    }
}

// ---------------------------------------------------------------------------
// Transpose V slice of qkv (bf16 [B*S, 3D], cols 2D..3D-1) -> Vt [B, D, S]
// ---------------------------------------------------------------------------
__global__ __launch_bounds__(256) void transpose_v(
    const __bf16* __restrict__ qkv, __bf16* __restrict__ Vt)
{
    __shared__ __bf16 tile[32][33];
    const int b  = blockIdx.z;
    const int s0 = blockIdx.x * 32;
    const int d0 = blockIdx.y * 32;
    #pragma unroll
    for (int j = 0; j < 4; ++j) {
        int s = s0 + threadIdx.y + j * 8;
        int d = d0 + threadIdx.x;
        tile[threadIdx.y + j * 8][threadIdx.x] =
            qkv[((size_t)(b * SEQ + s)) * (3 * DIMN) + 2 * DIMN + d];
    }
    __syncthreads();
    #pragma unroll
    for (int j = 0; j < 4; ++j) {
        int d = d0 + threadIdx.y + j * 8;
        int s = s0 + threadIdx.x;
        Vt[((size_t)(b * DIMN + d)) * SEQ + s] = tile[threadIdx.x][threadIdx.y + j * 8];
    }
}

// ---------------------------------------------------------------------------
// Row softmax over scores [SEQ, SEQ] f32 for one batch.
// masked_fill(-1e20) BEFORE 1/sqrt(D) scaling (matches reference), then
// numerically-stable softmax; emits bf16 probabilities.
// ---------------------------------------------------------------------------
__global__ __launch_bounds__(256) void softmax_rows(
    const float* __restrict__ scores, const int* __restrict__ mask,
    __bf16* __restrict__ probs)
{
    __shared__ float red[256];
    const int row = blockIdx.x;
    const int tid = threadIdx.x;
    const float* r = scores + (size_t)row * SEQ;
    const float scale = 0.03125f;  // 1/sqrt(1024)

    float vals[8];
    float mx = -3.0e38f;
    #pragma unroll
    for (int i = 0; i < 8; ++i) {
        int col = tid + i * 256;
        float v = r[col];
        if (mask[col] == 0) v = -1.0e20f;
        v *= scale;
        vals[i] = v;
        mx = fmaxf(mx, v);
    }
    red[tid] = mx;
    __syncthreads();
    for (int s = 128; s > 0; s >>= 1) {
        if (tid < s) red[tid] = fmaxf(red[tid], red[tid + s]);
        __syncthreads();
    }
    mx = red[0];
    __syncthreads();

    float sum = 0.0f;
    #pragma unroll
    for (int i = 0; i < 8; ++i) {
        vals[i] = __expf(vals[i] - mx);
        sum += vals[i];
    }
    red[tid] = sum;
    __syncthreads();
    for (int s = 128; s > 0; s >>= 1) {
        if (tid < s) red[tid] += red[tid + s];
        __syncthreads();
    }
    float inv = 1.0f / red[0];
    #pragma unroll
    for (int i = 0; i < 8; ++i) {
        int col = tid + i * 256;
        probs[(size_t)row * SEQ + col] = (__bf16)(vals[i] * inv);
    }
}

// ---------------------------------------------------------------------------
extern "C" void kernel_launch(void* const* d_in, const int* in_sizes, int n_in,
                              void* d_out, int out_size, void* d_ws, size_t ws_size,
                              hipStream_t stream)
{
    (void)in_sizes; (void)n_in; (void)out_size; (void)ws_size;
    const float* X     = (const float*)d_in[0];
    const int*   amask = (const int*)d_in[1];
    const float* W_in  = (const float*)d_in[2];
    const float* b_in  = (const float*)d_in[3];
    const float* W_out = (const float*)d_in[4];
    const float* b_out = (const float*)d_in[5];
    float* out = (float*)d_out;

    char* ws = (char*)d_ws;
    size_t off = 0;
    auto alloc = [&](size_t bytes) -> char* {
        char* p = ws + off;
        off += (bytes + 255) & ~(size_t)255;
        return p;
    };
    __bf16* Xbf    = (__bf16*)alloc((size_t)BATCH * SEQ * DIMN * 2);        // 16 MB
    __bf16* Wibf   = (__bf16*)alloc((size_t)3 * DIMN * DIMN * 2);           //  6 MB
    __bf16* Wobf   = (__bf16*)alloc((size_t)DIMN * DIMN * 2);               //  2 MB
    __bf16* qkvbf  = (__bf16*)alloc((size_t)BATCH * SEQ * 3 * DIMN * 2);    // 48 MB
    __bf16* Vt     = (__bf16*)alloc((size_t)BATCH * DIMN * SEQ * 2);        // 16 MB
    __bf16* Obf    = (__bf16*)alloc((size_t)BATCH * SEQ * DIMN * 2);        // 16 MB
    float*  scores = (float*)alloc((size_t)SEQ * SEQ * 4);                  // 16 MB
    __bf16* probs  = (__bf16*)alloc((size_t)SEQ * SEQ * 2);                 //  8 MB

    // --- f32 -> bf16 conversions ---
    {
        int n8 = BATCH * SEQ * DIMN / 8;
        cvt_f32_to_bf16<<<(n8 + 255) / 256, 256, 0, stream>>>(X, Xbf, n8);
        n8 = 3 * DIMN * DIMN / 8;
        cvt_f32_to_bf16<<<(n8 + 255) / 256, 256, 0, stream>>>(W_in, Wibf, n8);
        n8 = DIMN * DIMN / 8;
        cvt_f32_to_bf16<<<(n8 + 255) / 256, 256, 0, stream>>>(W_out, Wobf, n8);
    }

    // --- qkv = X @ W_in^T + b_in   [B*S, 3D] bf16 ---
    gemm_bf16_wmma<true, true><<<dim3(3 * DIMN / 128, BATCH * SEQ / 128), 256, 0, stream>>>(
        Xbf, DIMN, Wibf, DIMN, b_in, (void*)qkvbf, 3 * DIMN, DIMN);

    // --- Vt[b, d, s] = V[b, s, d] ---
    transpose_v<<<dim3(SEQ / 32, DIMN / 32, BATCH), dim3(32, 8), 0, stream>>>(qkvbf, Vt);

    // --- per-batch attention ---
    for (int b = 0; b < BATCH; ++b) {
        const __bf16* Q  = qkvbf + (size_t)b * SEQ * 3 * DIMN;
        const __bf16* Km = Q + DIMN;
        // scores = Q @ K^T   (f32 out)
        gemm_bf16_wmma<false, false><<<dim3(SEQ / 128, SEQ / 128), 256, 0, stream>>>(
            Q, 3 * DIMN, Km, 3 * DIMN, nullptr, (void*)scores, SEQ, DIMN);
        // softmax + mask + scale -> bf16 probs
        softmax_rows<<<SEQ, 256, 0, stream>>>(scores, amask + (size_t)b * SEQ, probs);
        // O = probs @ V = probs @ Vt^T   (bf16 out)
        gemm_bf16_wmma<true, false><<<dim3(DIMN / 128, SEQ / 128), 256, 0, stream>>>(
            probs, SEQ, Vt + (size_t)b * DIMN * SEQ, SEQ, nullptr,
            (void*)(Obf + (size_t)b * SEQ * DIMN), DIMN, SEQ);
    }

    // --- out = O @ W_out^T + b_out   (f32 out) ---
    gemm_bf16_wmma<false, true><<<dim3(DIMN / 128, BATCH * SEQ / 128), 256, 0, stream>>>(
        Obf, DIMN, Wobf, DIMN, b_out, (void*)out, DIMN, DIMN);
}
